// Gaussian_Orthogonal_MultiHeadSA_48000554500476
// MI455X (gfx1250) — matrix operand order repaired and checked by reading, not compiled, not run on hardware
//
#include <hip/hip_runtime.h>

typedef unsigned short u16;
typedef __attribute__((ext_vector_type(16))) __bf16 v16bf;
typedef __attribute__((ext_vector_type(8)))  float  v8f;

constexpr int kB = 2, kT = 1653, kD = 512, kDE = 64, kBE = 16, kTP = 1664;
constexpr int LDSP = 40;   // LDS row pitch in u16 units (80 B, 16B-aligned rows)

__device__ __forceinline__ u16 f2bf(float f) {
  union { float f; unsigned u; } x; x.f = f;
  unsigned u = x.u;
  unsigned r = u + 0x7FFFu + ((u >> 16) & 1u);   // round-to-nearest-even
  return (u16)(r >> 16);
}

union FragBF { v16bf v; uint4 q[2]; };

// ---- load a 128(row) x 32(col) tile into LDS (bf16), from f32 row-major src
__device__ __forceinline__ void ld_tile_f32(u16* dst, const float* src, long ld,
                                            int row0, int rlim, int col0, int clim,
                                            int tid) {
#pragma unroll
  for (int it = 0; it < 4; ++it) {
    int i = tid + it * 256;           // 1024 quads total
    int r = i >> 3;
    int c = (i & 7) << 2;
    int gr = row0 + r, gc = col0 + c;
    float f0 = 0.f, f1 = 0.f, f2 = 0.f, f3 = 0.f;
    if (gr < rlim) {
      const float* p = src + (long)gr * ld + gc;
      if (gc + 3 < clim) {
        f0 = p[0]; f1 = p[1]; f2 = p[2]; f3 = p[3];
        if (gc + 35 < clim) __builtin_prefetch(p + 32, 0, 3);  // next K tile
      } else {
        if (gc     < clim) f0 = p[0];
        if (gc + 1 < clim) f1 = p[1];
        if (gc + 2 < clim) f2 = p[2];
        if (gc + 3 < clim) f3 = p[3];
      }
    }
    uint2 w;
    w.x = (unsigned)f2bf(f0) | ((unsigned)f2bf(f1) << 16);
    w.y = (unsigned)f2bf(f2) | ((unsigned)f2bf(f3) << 16);
    *(uint2*)(dst + r * LDSP + c) = w;
  }
}

// ---- same, from bf16 (u16) row-major src
__device__ __forceinline__ void ld_tile_bf16(u16* dst, const u16* src, long ld,
                                             int row0, int rlim, int col0, int clim,
                                             int tid) {
#pragma unroll
  for (int it = 0; it < 4; ++it) {
    int i = tid + it * 256;
    int r = i >> 3;
    int c = (i & 7) << 2;
    int gr = row0 + r, gc = col0 + c;
    uint2 w; w.x = 0u; w.y = 0u;
    if (gr < rlim) {
      const u16* p = src + (long)gr * ld + gc;
      if (gc + 3 < clim) {
        w = *(const uint2*)p;
        if (gc + 35 < clim) __builtin_prefetch(p + 32, 0, 3);
      } else {
        u16 a0 = (gc     < clim) ? p[0] : (u16)0;
        u16 a1 = (gc + 1 < clim) ? p[1] : (u16)0;
        u16 a2 = (gc + 2 < clim) ? p[2] : (u16)0;
        u16 a3 = (gc + 3 < clim) ? p[3] : (u16)0;
        w.x = (unsigned)a0 | ((unsigned)a1 << 16);
        w.y = (unsigned)a2 | ((unsigned)a3 << 16);
      }
    }
    *(uint2*)(dst + r * LDSP + c) = w;
  }
}

// ---- V tile for att2@V: store transposed Bst[de][s], de<64, s-chunk of 32
__device__ __forceinline__ void ld_vT_tile(u16* dstB, const u16* Vsrc, int s0, int tid) {
#pragma unroll
  for (int it = 0; it < 2; ++it) {
    int i = tid + it * 256;           // 512 quads: 32 s-rows x 16 quads
    int s  = i >> 4;
    int de = (i & 15) << 2;
    int gs = s0 + s;
    u16 w0 = 0, w1 = 0, w2 = 0, w3 = 0;
    if (gs < kT) {
      uint2 q = *(const uint2*)(Vsrc + (long)gs * kDE + de);
      w0 = (u16)(q.x & 0xFFFFu); w1 = (u16)(q.x >> 16);
      w2 = (u16)(q.y & 0xFFFFu); w3 = (u16)(q.y >> 16);
    }
    dstB[(de + 0) * LDSP + s] = w0;
    dstB[(de + 1) * LDSP + s] = w1;
    dstB[(de + 2) * LDSP + s] = w2;
    dstB[(de + 3) * LDSP + s] = w3;
  }
}

// ---- one K-step: wave computes its 64x32 subtile (4x2 wmma)
__device__ __forceinline__ void mma_8(const u16* As, const u16* Bs, int m0, int n0,
                                      int lane, v8f c[4][2]) {
  int m = lane & 15, hi = lane >> 4;
  FragBF a[4], b[2];
#pragma unroll
  for (int mi = 0; mi < 4; ++mi) {
    const u16* rp = As + (m0 + mi * 16 + m) * LDSP;
    a[mi].q[0] = *(const uint4*)(rp + hi * 8);        // K = hi*8 .. +7
    a[mi].q[1] = *(const uint4*)(rp + 16 + hi * 8);   // K = 16+hi*8 .. +7
  }
#pragma unroll
  for (int ni = 0; ni < 2; ++ni) {
    const u16* rp = Bs + (n0 + ni * 16 + m) * LDSP;   // Bst[n][k], k-contiguous
    b[ni].q[0] = *(const uint4*)(rp + hi * 8);        // K = hi*8 .. +7
    b[ni].q[1] = *(const uint4*)(rp + 16 + hi * 8);   // K = 16+hi*8 .. +7
  }
#pragma unroll
  for (int mi = 0; mi < 4; ++mi)
#pragma unroll
    for (int ni = 0; ni < 2; ++ni)
      c[mi][ni] = __builtin_amdgcn_wmma_f32_16x16x32_bf16(
          false, a[mi].v, false, b[ni].v, (short)0, c[mi][ni], false, false);
}

// ============ Kernel 1: Q/K/V projection, transposed layout [B][D][T] ============
// C[e][t] = sum_d W[e][d] * X[t][d] + b[e]   (+ learnable sigmoid for q,k)
__global__ void __launch_bounds__(256) k_proj(
    const float* __restrict__ X,
    const float* __restrict__ Wq, const float* __restrict__ bq,
    const float* __restrict__ Wk, const float* __restrict__ bk,
    const float* __restrict__ Wv, const float* __restrict__ bv,
    u16* __restrict__ Qh, u16* __restrict__ Kh, u16* __restrict__ Vh) {
  __shared__ u16 As[128 * LDSP];
  __shared__ u16 Bs[128 * LDSP];
  int tid = threadIdx.x, lane = tid & 31, wid = tid >> 5;
  int b = blockIdx.z / 3, pr = blockIdx.z % 3;
  const float* W  = pr == 0 ? Wq : (pr == 1 ? Wk : Wv);
  const float* bb = pr == 0 ? bq : (pr == 1 ? bk : bv);
  u16* dst        = pr == 0 ? Qh : (pr == 1 ? Kh : Vh);
  const float* Xb = X + (long)b * kT * kD;
  int tM0 = blockIdx.y * 128;   // e
  int tN0 = blockIdx.x * 128;   // t
  v8f c[4][2] = {};
  for (int k0 = 0; k0 < kD; k0 += 32) {
    ld_tile_f32(As, W,  kD, tM0, kD, k0, kD, tid);
    ld_tile_f32(Bs, Xb, kD, tN0, kT, k0, kD, tid);
    __syncthreads();
    mma_8(As, Bs, (wid >> 2) * 64, (wid & 3) * 32, lane, c);
    __syncthreads();
  }
  int m = lane & 15, hi = lane >> 4;
  int m0w = (wid >> 2) * 64, n0w = (wid & 3) * 32;
#pragma unroll
  for (int mi = 0; mi < 4; ++mi)
#pragma unroll
    for (int ni = 0; ni < 2; ++ni) {
      int e0 = tM0 + m0w + mi * 16 + hi * 8;
      int t  = tN0 + n0w + ni * 16 + m;
      if (t >= kT) continue;
#pragma unroll
      for (int v = 0; v < 8; ++v) {
        int e = e0 + v;
        float val = c[mi][ni][v] + bb[e];
        if (pr < 2) val = 1.2f / (1.0f + __expf(-1.6f * val));
        dst[((long)b * kD + e) * kT + t] = f2bf(val);   // [B][D][T] == [BE][T][64]
      }
    }
}

// ============ Kernel 2: attP = (Q·Kᵀ) * T^-1/2 * punish,  bf16 [BE][T][TP] ============
__global__ void __launch_bounds__(256) k_qkt(
    const u16* __restrict__ Qh, const u16* __restrict__ Kh,
    const float* __restrict__ punish, u16* __restrict__ attP) {
  __shared__ u16 As[128 * LDSP];
  __shared__ u16 Bs[128 * LDSP];
  int tid = threadIdx.x, lane = tid & 31, wid = tid >> 5;
  int be = blockIdx.z;
  const u16* Q = Qh + (long)be * kT * kDE;
  const u16* K = Kh + (long)be * kT * kDE;
  int tM0 = blockIdx.y * 128;   // t
  int tN0 = blockIdx.x * 128;   // s
  v8f c[4][2] = {};
  for (int k0 = 0; k0 < kDE; k0 += 32) {
    ld_tile_bf16(As, Q, kDE, tM0, kT, k0, kDE, tid);
    ld_tile_bf16(Bs, K, kDE, tN0, kT, k0, kDE, tid);
    __syncthreads();
    mma_8(As, Bs, (wid >> 2) * 64, (wid & 3) * 32, lane, c);
    __syncthreads();
  }
  float inv = rsqrtf((float)kT);
  int m = lane & 15, hi = lane >> 4;
  int m0w = (wid >> 2) * 64, n0w = (wid & 3) * 32;
#pragma unroll
  for (int mi = 0; mi < 4; ++mi)
#pragma unroll
    for (int ni = 0; ni < 2; ++ni) {
      int t0 = tM0 + m0w + mi * 16 + hi * 8;
      int s  = tN0 + n0w + ni * 16 + m;
#pragma unroll
      for (int v = 0; v < 8; ++v) {
        int t = t0 + v;
        if (t >= kT) continue;
        float val = 0.f;
        if (s < kT) val = c[mi][ni][v] * inv * punish[(long)t * kT + s];
        attP[((long)be * kT + t) * kTP + s] = f2bf(val);   // pad cols written as 0
      }
    }
}

// ============ Kernel 3: att2 = attP · orthᵀ, fused mask(==0 -> -2e20) ============
__global__ void __launch_bounds__(256) k_orth(
    const u16* __restrict__ attP, const float* __restrict__ orth,
    u16* __restrict__ att2) {
  __shared__ u16 As[128 * LDSP];
  __shared__ u16 Bs[128 * LDSP];
  int tid = threadIdx.x, lane = tid & 31, wid = tid >> 5;
  int be = blockIdx.z;
  const u16* A   = attP + (long)be * kT * kTP;
  const float* O = orth + (long)be * kT * kT;
  int tM0 = blockIdx.y * 128;   // t
  int tN0 = blockIdx.x * 128;   // s
  v8f c[4][2] = {};
  for (int k0 = 0; k0 < kTP; k0 += 32) {
    ld_tile_bf16(As, A, kTP, tM0, kT, k0, kTP, tid);
    ld_tile_f32 (Bs, O, kT,  tN0, kT, k0, kT,  tid);
    __syncthreads();
    mma_8(As, Bs, (wid >> 2) * 64, (wid & 3) * 32, lane, c);
    __syncthreads();
  }
  int m = lane & 15, hi = lane >> 4;
  int m0w = (wid >> 2) * 64, n0w = (wid & 3) * 32;
#pragma unroll
  for (int mi = 0; mi < 4; ++mi)
#pragma unroll
    for (int ni = 0; ni < 2; ++ni) {
      int t0 = tM0 + m0w + mi * 16 + hi * 8;
      int s  = tN0 + n0w + ni * 16 + m;
#pragma unroll
      for (int v = 0; v < 8; ++v) {
        int t = t0 + v;
        if (t >= kT) continue;
        float val = 0.f;
        if (s < kT) {
          float a = c[mi][ni][v];
          val = (a == 0.0f) ? -2e20f : a;       // masked_fill(att==0, -2e20)
        }
        att2[((long)be * kT + t) * kTP + s] = f2bf(val);
      }
    }
}

// ============ Kernel 4: out = att2 · V, stored head-concat into [B][T][D] ============
__global__ void __launch_bounds__(256) k_av(
    const u16* __restrict__ att2, const u16* __restrict__ Vh,
    u16* __restrict__ outB) {
  __shared__ u16 As[128 * LDSP];
  __shared__ u16 Bs[128 * LDSP];
  int tid = threadIdx.x, lane = tid & 31, wid = tid >> 5;
  int be = blockIdx.z;
  int b = be >> 3, e = be & 7;
  const u16* A = att2 + (long)be * kT * kTP;
  const u16* V = Vh   + (long)be * kT * kDE;
  int tM0 = blockIdx.y * 128;   // t  (N dim is just 64 here)
  // zero the unused B rows (n = 64..127) once
  for (int i = tid; i < 64 * LDSP; i += 256) Bs[64 * LDSP + i] = 0;
  v8f c[4][2] = {};
  for (int k0 = 0; k0 < kTP; k0 += 32) {
    ld_tile_bf16(As, A, kTP, tM0, kT, k0, kTP, tid);
    ld_vT_tile(Bs, V, k0, tid);
    __syncthreads();
    mma_8(As, Bs, (wid >> 2) * 64, (wid & 3) * 32, lane, c);
    __syncthreads();
  }
  int m = lane & 15, hi = lane >> 4;
  int m0w = (wid >> 2) * 64, n0w = (wid & 3) * 32;
#pragma unroll
  for (int mi = 0; mi < 4; ++mi)
#pragma unroll
    for (int ni = 0; ni < 2; ++ni) {
      int t0 = tM0 + m0w + mi * 16 + hi * 8;
      int de = n0w + ni * 16 + m;
      if (de >= kDE) continue;
#pragma unroll
      for (int v = 0; v < 8; ++v) {
        int t = t0 + v;
        if (t >= kT) continue;
        outB[((long)(b * kT + t)) * kD + e * kDE + de] = f2bf(c[mi][ni][v]);
      }
    }
}

// ============ Kernel 5: final projection  out = X2 · Woᵀ + bo  (f32 output) ============
__global__ void __launch_bounds__(256) k_outproj(
    const u16* __restrict__ outB, const float* __restrict__ Wo,
    const float* __restrict__ bo, float* __restrict__ out) {
  __shared__ u16 As[128 * LDSP];
  __shared__ u16 Bs[128 * LDSP];
  int tid = threadIdx.x, lane = tid & 31, wid = tid >> 5;
  int b = blockIdx.z;
  const u16* A = outB + (long)b * kT * kD;
  int tM0 = blockIdx.y * 128;   // t
  int tN0 = blockIdx.x * 128;   // e'
  v8f c[4][2] = {};
  for (int k0 = 0; k0 < kD; k0 += 32) {
    ld_tile_bf16(As, A,  kD, tM0, kT, k0, kD, tid);
    ld_tile_f32 (Bs, Wo, kD, tN0, kD, k0, kD, tid);
    __syncthreads();
    mma_8(As, Bs, (wid >> 2) * 64, (wid & 3) * 32, lane, c);
    __syncthreads();
  }
  int m = lane & 15, hi = lane >> 4;
  int m0w = (wid >> 2) * 64, n0w = (wid & 3) * 32;
#pragma unroll
  for (int mi = 0; mi < 4; ++mi)
#pragma unroll
    for (int ni = 0; ni < 2; ++ni) {
      int t0 = tM0 + m0w + mi * 16 + hi * 8;
      int n  = tN0 + n0w + ni * 16 + m;
#pragma unroll
      for (int v = 0; v < 8; ++v) {
        int t = t0 + v;
        if (t >= kT) continue;
        out[((long)b * kT + t) * kD + n] = c[mi][ni][v] + bo[n];
      }
    }
}

extern "C" void kernel_launch(void* const* d_in, const int* in_sizes, int n_in,
                              void* d_out, int out_size, void* d_ws, size_t ws_size,
                              hipStream_t stream) {
  (void)in_sizes; (void)n_in; (void)out_size; (void)ws_size;
  const float* X      = (const float*)d_in[0];
  const float* Wq     = (const float*)d_in[1];
  const float* bq     = (const float*)d_in[2];
  const float* Wk     = (const float*)d_in[3];
  const float* bk     = (const float*)d_in[4];
  const float* Wv     = (const float*)d_in[5];
  const float* bv     = (const float*)d_in[6];
  const float* Wo     = (const float*)d_in[7];
  const float* bo     = (const float*)d_in[8];
  const float* punish = (const float*)d_in[9];
  const float* orth   = (const float*)d_in[10];
  float* out = (float*)d_out;

  char* w = (char*)d_ws;
  size_t szHead = (size_t)kBE * kT * kDE * sizeof(u16);   // ~3.4 MB
  size_t szAtt  = (size_t)kBE * kT * kTP * sizeof(u16);   // ~88 MB
  u16* Qh   = (u16*)w; w += szHead;
  u16* Kh   = (u16*)w; w += szHead;
  u16* Vh   = (u16*)w; w += szHead;
  u16* attP = (u16*)w; w += szAtt;
  u16* att2 = (u16*)w; w += szAtt;
  u16* outB = (u16*)w;

  dim3 blk(256);
  int nT = (kT + 127) / 128;   // 13 (TP/128 == 13 as well)
  k_proj   <<<dim3(nT, kD / 128, kB * 3), blk, 0, stream>>>(X, Wq, bq, Wk, bk, Wv, bv, Qh, Kh, Vh);
  k_qkt    <<<dim3(nT, nT, kBE),          blk, 0, stream>>>(Qh, Kh, punish, attP);
  k_orth   <<<dim3(nT, nT, kBE),          blk, 0, stream>>>(attP, orth, att2);
  k_av     <<<dim3(1,  nT, kBE),          blk, 0, stream>>>(att2, Vh, outB);
  k_outproj<<<dim3(kD / 128, nT, kB),     blk, 0, stream>>>(outB, Wo, bo, out);
}
